// LatentSDE_34729105555640
// MI455X (gfx1250) — matrix operand order repaired and checked by reading, not compiled, not run on hardware
//
#include <hip/hip_runtime.h>
#include <hip/hip_bf16.h>

// ---------------------------------------------------------------------------
// Types for CDNA5 WMMA (wave32): v_wmma_f32_16x16x32_f16
// ---------------------------------------------------------------------------
typedef _Float16 v16h __attribute__((ext_vector_type(16)));
typedef _Float16 v8h  __attribute__((ext_vector_type(8)));
typedef float    v8f  __attribute__((ext_vector_type(8)));

static __device__ __forceinline__ float sigmoid_f(float x) {
    return 1.0f / (1.0f + __expf(-x));
}

// ---------------------------------------------------------------------------
// Elementwise f32 -> f16 convert
// ---------------------------------------------------------------------------
__global__ __launch_bounds__(256) void cvt_kernel(const float* __restrict__ src,
                                                  _Float16* __restrict__ dst, int n) {
    for (int i = blockIdx.x * blockDim.x + threadIdx.x; i < n;
         i += gridDim.x * blockDim.x)
        dst[i] = (_Float16)src[i];
}

// f32 (rows,cols) -> f16 transposed (cols,rows): produces B-operand layout (K,N)
__global__ __launch_bounds__(256) void cvtT_kernel(const float* __restrict__ src,
                                                   _Float16* __restrict__ dst,
                                                   int rows, int cols) {
    int total = rows * cols;
    for (int i = blockIdx.x * blockDim.x + threadIdx.x; i < total;
         i += gridDim.x * blockDim.x) {
        int r = i / cols, c = i - r * cols;
        dst[(size_t)c * rows + r] = (_Float16)src[i];
    }
}

__global__ void zero_kernel(float* p) { if (threadIdx.x == 0) p[0] = 0.0f; }

// ---------------------------------------------------------------------------
// WMMA GEMM: Out[M,N] = X[M,K] * Wt[K,N] + bias[N]   (optionally relu, f16 out)
// X row-major f16, Wt row-major f16 (already transposed weight).
// One 16x16 tile per wave; 8 waves (256 thr) per block cover 128 columns.
// Fragment layouts per CDNA5 ISA 7.12.2 (16-bit A 16x32; B lane<->K; C/D table).
// ---------------------------------------------------------------------------
template <bool RELU, bool HALF_OUT>
__global__ __launch_bounds__(256) void wmma_gemm_kernel(
    const _Float16* __restrict__ X, const _Float16* __restrict__ Wt,
    const float* __restrict__ bias, void* __restrict__ Out,
    int M, int N, int K) {
    const int lane = threadIdx.x & 31;
    const int wave = threadIdx.x >> 5;
    const int m0 = blockIdx.x * 16;
    const int n0 = (blockIdx.y * 8 + wave) * 16;
    if (n0 >= N) return;                 // wave-uniform: EXEC stays all-ones
    const int g   = lane >> 4;           // lane group (0/1)
    const int r16 = lane & 15;

    const _Float16* xrow = X + (size_t)(m0 + r16) * K + 8 * g;  // A: row m, K-offset 8g
    v8f acc = {};
    for (int k0 = 0; k0 < K; k0 += 32) {
        __builtin_prefetch(xrow + k0 + 128, 0, 1);   // global_prefetch_b8
        v8h a_lo = *(const v8h*)(xrow + k0);         // K = k0+8g .. +7
        v8h a_hi = *(const v8h*)(xrow + k0 + 16);    // K = k0+16+8g .. +7
        v16h a = __builtin_shufflevector(a_lo, a_hi, 0, 1, 2, 3, 4, 5, 6, 7,
                                         8, 9, 10, 11, 12, 13, 14, 15);
        // B: lane l holds column N = n0 + l%16, K = k0+8g..+7 and k0+16+8g..+7
        v16h b;
        for (int s = 0; s < 16; ++s)
            b[s] = Wt[(size_t)(k0 + (s < 8 ? 8 * g + s : 16 + 8 * g + (s - 8))) * N + n0 + r16];
        acc = __builtin_amdgcn_wmma_f32_16x16x32_f16(
            /*neg_a=*/false, a, /*neg_b=*/false, b,
            /*c_mod=*/(short)0, acc, /*reuse_a=*/false, /*reuse_b=*/false);
    }
    const int n = n0 + r16;
    const float bn = bias ? bias[n] : 0.0f;
#pragma unroll
    for (int r = 0; r < 8; ++r) {        // D: VGPR r -> m = r + 8g, lane -> n
        const int m = m0 + r + 8 * g;
        float v = acc[r] + bn;
        if (RELU) v = v > 0.0f ? v : 0.0f;
        if (HALF_OUT)
            ((_Float16*)Out)[(size_t)m * N + n] = (_Float16)v;
        else
            ((float*)Out)[(size_t)m * N + n] = v;
    }
}

// ---------------------------------------------------------------------------
// Serial GRU over reversed time. GI = xs @ Wih^T + bih precomputed (T,768).
// hs stored at forward index (T-1-s) so ctx GEMM needs no flip.
// One block, 256 threads (H=256); h lives in LDS; Whh streams from L2.
// ---------------------------------------------------------------------------
__global__ __launch_bounds__(256) void gru_kernel(
    const float* __restrict__ GI, const float* __restrict__ Whh,
    const float* __restrict__ bhh, float* __restrict__ hs, int T) {
    __shared__ float h[256];
    const int tid = threadIdx.x;
    h[tid] = 0.0f;
    __syncthreads();
    const float* w_r = Whh + (size_t)tid * 256;
    const float* w_z = Whh + (size_t)(tid + 256) * 256;
    const float* w_g = Whh + (size_t)(tid + 512) * 256;
    const float b_r = bhh[tid], b_z = bhh[tid + 256], b_g = bhh[tid + 512];
    for (int s = 0; s < T; ++s) {
        const int tr = T - 1 - s;
        float ar = b_r, az = b_z, ag = b_g;
#pragma unroll 8
        for (int k = 0; k < 256; ++k) {
            const float hk = h[k];
            ar = fmaf(w_r[k], hk, ar);
            az = fmaf(w_z[k], hk, az);
            ag = fmaf(w_g[k], hk, ag);
        }
        const float* gi = GI + (size_t)tr * 768;
        const float rg = sigmoid_f(gi[tid] + ar);
        const float zg = sigmoid_f(gi[256 + tid] + az);
        const float ng = tanhf(gi[512 + tid] + rg * ag);
        const float hn = (1.0f - zg) * ng + zg * h[tid];
        __syncthreads();
        h[tid] = hn;
        hs[(size_t)tr * 256 + tid] = hn;
        __syncthreads();
    }
}

// ---------------------------------------------------------------------------
// Serial Euler-Maruyama SDE + KL accumulators. One block, 256 threads.
// L=32 latent, K=32 control, H=256 hidden, C=128 context.
// Writes zs (T,32), out[1]=logqp0, out[2]=logqp_path.
// ---------------------------------------------------------------------------
__global__ __launch_bounds__(256) void sde_kernel(
    const float* __restrict__ ts, const float* __restrict__ z0_eps,
    const float* __restrict__ dWn, const float* __restrict__ ctx,
    const float* __restrict__ qW, const float* __restrict__ qb,
    const float* __restrict__ fW1, const float* __restrict__ fb1,
    const float* __restrict__ fW2, const float* __restrict__ fb2,
    const float* __restrict__ hW1, const float* __restrict__ hb1,
    const float* __restrict__ hW2, const float* __restrict__ hb2,
    const float* __restrict__ gW1, const float* __restrict__ gb1,
    const float* __restrict__ gW2, const float* __restrict__ gb2,
    const float* __restrict__ pz0_mean, const float* __restrict__ pz0_logvar,
    float* __restrict__ zs, float* __restrict__ out, int T) {
    __shared__ float zsh[32], csh[128], h1[256], Gm[1024], Am[32 * 33];
    __shared__ float rhs[32], delta[32], uvec[32], zfv[32], red[32];
    __shared__ float qz0m[32], qz0ls[32], kl;
    const int tid = threadIdx.x;

    // ---- z0 from posterior + KL(q||p) at t=0 ----
    if (tid < 128) csh[tid] = ctx[tid];
    __syncthreads();
    if (tid < 64) {
        float acc = qb[tid];
        for (int k = 0; k < 128; ++k) acc = fmaf(qW[tid * 128 + k], csh[k], acc);
        if (tid < 32) qz0m[tid] = acc;
        else qz0ls[tid - 32] = fminf(fmaxf(acc, -5.0f), 2.0f);
    }
    __syncthreads();
    if (tid < 32) {
        const float zl = qz0m[tid] + __expf(qz0ls[tid]) * z0_eps[tid];
        zsh[tid] = zl;
        zs[tid] = zl;
        const float pls = fminf(fmaxf(0.5f * pz0_logvar[tid], -5.0f), 2.0f);
        const float qls = qz0ls[tid];
        const float dm = qz0m[tid] - pz0_mean[tid];
        red[tid] = pls - qls +
                   (__expf(2.0f * qls) + dm * dm) / (2.0f * __expf(2.0f * pls)) -
                   0.5f;
    }
    __syncthreads();
    if (tid == 0) {
        float s = 0.0f;
        for (int i = 0; i < 32; ++i) s += red[i];
        out[1] = s;
        kl = 0.0f;
    }
    __syncthreads();

    // ---- Euler-Maruyama loop ----
    for (int t = 0; t < T - 1; ++t) {
        const float tt = ts[t];
        const float dt = ts[t + 1] - tt;
        if (tid < 128) csh[tid] = ctx[(size_t)t * 128 + tid];
        __syncthreads();
        // f hidden: fW1 (256 x 161) on [c(128), t, z(32)]
        {
            const float* w = fW1 + (size_t)tid * 161;
            float acc = fb1[tid];
            for (int k = 0; k < 128; ++k) acc = fmaf(w[k], csh[k], acc);
            acc = fmaf(w[128], tt, acc);
            for (int j = 0; j < 32; ++j) acc = fmaf(w[129 + j], zsh[j], acc);
            h1[tid] = tanhf(acc);
        }
        __syncthreads();
        if (tid < 32) {
            const float* w = fW2 + (size_t)tid * 256;
            float acc = fb2[tid];
            for (int k = 0; k < 256; ++k) acc = fmaf(w[k], h1[k], acc);
            zfv[tid] = acc;
        }
        __syncthreads();
        // h hidden: hW1 (256 x 33) on [t, z]
        {
            const float* w = hW1 + (size_t)tid * 33;
            float acc = fmaf(w[0], tt, hb1[tid]);
            for (int j = 0; j < 32; ++j) acc = fmaf(w[1 + j], zsh[j], acc);
            h1[tid] = tanhf(acc);
        }
        __syncthreads();
        if (tid < 32) {
            const float* w = hW2 + (size_t)tid * 256;
            float acc = hb2[tid];
            for (int k = 0; k < 256; ++k) acc = fmaf(w[k], h1[k], acc);
            delta[tid] = zfv[tid] - acc;
        }
        __syncthreads();
        // g hidden: gW1 (256 x 33) on [t, z]
        {
            const float* w = gW1 + (size_t)tid * 33;
            float acc = fmaf(w[0], tt, gb1[tid]);
            for (int j = 0; j < 32; ++j) acc = fmaf(w[1 + j], zsh[j], acc);
            h1[tid] = tanhf(acc);
        }
        __syncthreads();
        // G flat (L*K = 1024) = gW2 @ h1 + gb2 ; 4 outputs per thread
#pragma unroll
        for (int q = 0; q < 4; ++q) {
            const int o = tid + 256 * q;
            const float* w = gW2 + (size_t)o * 256;
            float acc = gb2[o];
            for (int k = 0; k < 256; ++k) acc = fmaf(w[k], h1[k], acc);
            Gm[o] = acc;
        }
        __syncthreads();
        // A = G^T G + 1e-6 I  (32x32, ctrl dims); rhs = G^T delta
#pragma unroll
        for (int q = 0; q < 4; ++q) {
            const int idx = tid + 256 * q;
            const int i = idx >> 5, j = idx & 31;
            float acc = (i == j) ? 1e-6f : 0.0f;
            for (int l = 0; l < 32; ++l)
                acc = fmaf(Gm[l * 32 + i], Gm[l * 32 + j], acc);
            Am[i * 33 + j] = acc;
        }
        if (tid < 32) {
            float acc = 0.0f;
            for (int l = 0; l < 32; ++l) acc = fmaf(Gm[l * 32 + tid], delta[l], acc);
            rhs[tid] = acc;
        }
        __syncthreads();
        // Row-parallel Gaussian elimination (SPD, no pivoting)
        for (int k = 0; k < 31; ++k) {
            if (tid > k && tid < 32) {
                const float f = Am[tid * 33 + k] / Am[k * 33 + k];
                for (int j = k; j < 32; ++j) Am[tid * 33 + j] -= f * Am[k * 33 + j];
                rhs[tid] -= f * rhs[k];
            }
            __syncthreads();
        }
        // Back-substitution + KL rate on thread 0
        if (tid == 0) {
            float uu = 0.0f;
            for (int i = 31; i >= 0; --i) {
                float s = rhs[i];
                for (int j = i + 1; j < 32; ++j) s = fmaf(-Am[i * 33 + j], uvec[j], s);
                const float ui = s / Am[i * 33 + i];
                uvec[i] = ui;
                uu = fmaf(ui, ui, uu);
            }
            kl += 0.5f * uu * dt;
        }
        // z_{t+1} = z + zf*dt + G @ (sqrt(dt) * w)
        const float sdt = sqrtf(dt);
        if (tid < 32) {
            float acc = zsh[tid] + zfv[tid] * dt;
            const float* wrow = dWn + (size_t)t * 32;
            for (int k = 0; k < 32; ++k)
                acc = fmaf(Gm[tid * 32 + k] * sdt, wrow[k], acc);
            zs[(size_t)(t + 1) * 32 + tid] = acc;
            red[tid] = acc;
        }
        __syncthreads();
        if (tid < 32) zsh[tid] = red[tid];
        __syncthreads();
    }
    if (tid == 0) out[2] = kl;
}

// ---------------------------------------------------------------------------
// Gaussian log-likelihood reduction: out[0] += sum(logpdf)/T via f32 atomics
// ---------------------------------------------------------------------------
__global__ __launch_bounds__(256) void loglik_kernel(
    const float* __restrict__ xs, const float* __restrict__ xhat,
    const float* __restrict__ log_sigma, float* __restrict__ out0,
    int total, int D, float invT) {
    __shared__ float red[256];
    float s = 0.0f;
    for (int i = blockIdx.x * blockDim.x + threadIdx.x; i < total;
         i += gridDim.x * blockDim.x) {
        const int d = i % D;
        const float ls = fminf(fmaxf(log_sigma[d], -4.0f), 2.0f);
        const float sig = __expf(ls);
        const float r = (xs[i] - xhat[i]) / sig;
        s += -0.5f * r * r - ls - 0.9189385332046727f;  // 0.5*log(2*pi)
    }
    red[threadIdx.x] = s;
    __syncthreads();
    for (int off = 128; off > 0; off >>= 1) {
        if (threadIdx.x < off) red[threadIdx.x] += red[threadIdx.x + off];
        __syncthreads();
    }
    if (threadIdx.x == 0) atomicAdd(out0, red[0] * invT);
}

// ---------------------------------------------------------------------------
// Host orchestration
// ---------------------------------------------------------------------------
extern "C" void kernel_launch(void* const* d_in, const int* in_sizes, int n_in,
                              void* d_out, int out_size, void* d_ws, size_t ws_size,
                              hipStream_t stream) {
    (void)n_in; (void)out_size; (void)ws_size;
    // setup_inputs order
    const float* xs       = (const float*)d_in[0];
    const float* ts       = (const float*)d_in[1];
    const float* z0_eps   = (const float*)d_in[2];
    const float* dWn      = (const float*)d_in[3];
    const float* enc_Wih  = (const float*)d_in[4];
    const float* enc_Whh  = (const float*)d_in[5];
    const float* enc_bih  = (const float*)d_in[6];
    const float* enc_bhh  = (const float*)d_in[7];
    const float* enc_Wc   = (const float*)d_in[8];
    const float* enc_bc   = (const float*)d_in[9];
    const float* qW       = (const float*)d_in[10];
    const float* qb       = (const float*)d_in[11];
    const float* fW1      = (const float*)d_in[12];
    const float* fb1      = (const float*)d_in[13];
    const float* fW2      = (const float*)d_in[14];
    const float* fb2      = (const float*)d_in[15];
    const float* hW1      = (const float*)d_in[16];
    const float* hb1      = (const float*)d_in[17];
    const float* hW2      = (const float*)d_in[18];
    const float* hb2      = (const float*)d_in[19];
    const float* gW1      = (const float*)d_in[20];
    const float* gb1      = (const float*)d_in[21];
    const float* gW2      = (const float*)d_in[22];
    const float* gb2      = (const float*)d_in[23];
    const float* decW1    = (const float*)d_in[24];
    const float* decb1    = (const float*)d_in[25];
    const float* decW2    = (const float*)d_in[26];
    const float* decb2    = (const float*)d_in[27];
    const float* pz0_mean = (const float*)d_in[28];
    const float* pz0_lv   = (const float*)d_in[29];
    const float* log_sig  = (const float*)d_in[30];
    float* out = (float*)d_out;

    const int T = in_sizes[1];         // 32768
    const int D = 128, L = 32, H = 256;
    const int G3 = 3 * H;              // 768

    // workspace carve-out (256B aligned)
    char* w = (char*)d_ws;
    size_t off = 0;
    auto carve = [&](size_t bytes) -> void* {
        void* p = w + off;
        off = (off + bytes + 255) & ~(size_t)255;
        return p;
    };
    _Float16* xs_h  = (_Float16*)carve((size_t)T * D * 2);
    _Float16* WihT  = (_Float16*)carve((size_t)G3 * D * 2);   // (D, 3H)
    float*    GI    = (float*)   carve((size_t)T * G3 * 4);
    float*    hs    = (float*)   carve((size_t)T * H * 4);
    _Float16* hs_h  = (_Float16*)carve((size_t)T * H * 2);
    _Float16* WcT   = (_Float16*)carve((size_t)H * D * 2);    // (H, C)
    float*    ctx   = (float*)   carve((size_t)T * D * 4);
    float*    zsb   = (float*)   carve((size_t)T * L * 4);
    _Float16* zs_h  = (_Float16*)carve((size_t)T * L * 2);
    _Float16* W1T   = (_Float16*)carve((size_t)H * L * 2);    // (L, H)
    _Float16* A1_h  = (_Float16*)carve((size_t)T * H * 2);
    _Float16* W2T   = (_Float16*)carve((size_t)D * H * 2);    // (H, D)
    float*    xhat  = (float*)   carve((size_t)T * D * 4);

    zero_kernel<<<1, 32, 0, stream>>>(out);

    // precision conversion + weight transposition into B-operand (K,N) layout
    cvt_kernel <<<1024, 256, 0, stream>>>(xs, xs_h, T * D);
    cvtT_kernel<<<64,   256, 0, stream>>>(enc_Wih, WihT, G3, D);  // (768,128)->(128,768)
    cvtT_kernel<<<64,   256, 0, stream>>>(enc_Wc,  WcT,  D,  H);  // (128,256)->(256,128)
    cvtT_kernel<<<8,    256, 0, stream>>>(decW1,   W1T,  H,  L);  // (256,32) ->(32,256)
    cvtT_kernel<<<64,   256, 0, stream>>>(decW2,   W2T,  D,  H);  // (128,256)->(256,128)

    // GI = xs @ Wih^T + bih : M=T, N=768, K=128
    wmma_gemm_kernel<false, false><<<dim3(T / 16, G3 / 128), 256, 0, stream>>>(
        xs_h, WihT, enc_bih, GI, T, G3, D);

    // serial GRU over reversed time (stores hs in forward order)
    gru_kernel<<<1, 256, 0, stream>>>(GI, enc_Whh, enc_bhh, hs, T);

    cvt_kernel<<<1024, 256, 0, stream>>>(hs, hs_h, T * H);

    // ctx = hs @ Wc^T + bc : M=T, N=128, K=256
    wmma_gemm_kernel<false, false><<<dim3(T / 16, 1), 256, 0, stream>>>(
        hs_h, WcT, enc_bc, ctx, T, D, H);

    // serial SDE loop (writes zs, out[1], out[2])
    sde_kernel<<<1, 256, 0, stream>>>(ts, z0_eps, dWn, ctx, qW, qb,
                                      fW1, fb1, fW2, fb2, hW1, hb1, hW2, hb2,
                                      gW1, gb1, gW2, gb2, pz0_mean, pz0_lv,
                                      zsb, out, T);

    cvt_kernel<<<256, 256, 0, stream>>>(zsb, zs_h, T * L);

    // decoder: A1 = relu(zs @ decW1^T + b1) : M=T, N=256, K=32 (f16 out)
    wmma_gemm_kernel<true, true><<<dim3(T / 16, 2), 256, 0, stream>>>(
        zs_h, W1T, decb1, A1_h, T, H, L);
    // xhat = A1 @ decW2^T + b2 : M=T, N=128, K=256
    wmma_gemm_kernel<false, false><<<dim3(T / 16, 1), 256, 0, stream>>>(
        A1_h, W2T, decb2, xhat, T, D, H);

    // log_pxs = mean_t sum_d logpdf  -> out[0]
    loglik_kernel<<<1024, 256, 0, stream>>>(xs, xhat, log_sig, out,
                                            T * D, D, 1.0f / (float)T);
}
